// Bottleneck_61022895341823
// MI455X (gfx1250) — 8-bit matrix operand order repaired and checked by reading, compiled, not run on hardware
//
#include <hip/hip_runtime.h>
#include <stdint.h>

typedef int v8i __attribute__((ext_vector_type(8)));
typedef int v4i_vec __attribute__((vector_size(16)));
typedef __attribute__((address_space(1))) int8_t g_i8;
typedef __attribute__((address_space(3))) int8_t l_i8;
typedef __attribute__((address_space(1))) v4i_vec g_v4i;
typedef __attribute__((address_space(3))) v4i_vec l_v4i;

#if defined(__has_builtin)
#if __has_builtin(__builtin_amdgcn_global_load_async_to_lds_b128)
#define USE_ASYNC_LDS 1
#endif
#endif
#ifndef USE_ASYNC_LDS
#define USE_ASYNC_LDS 0
#endif

#if USE_ASYNC_LDS
#if __has_builtin(__builtin_amdgcn_s_wait_asynccnt)
#define WAIT_ASYNC(n) __builtin_amdgcn_s_wait_asynccnt(n)
#else
#define WAIT_ASYNC(n) asm volatile("s_wait_asynccnt %0" ::"i"(n) : "memory")
#endif
#endif

enum { SL_X=0, SL_W1, SL_B1, SL_W2, SL_B2, SL_W3, SL_B3, SL_WS, SL_BS, SL_A1, SL_A2, SL_A3, SL_COUNT };

__device__ __forceinline__ float slot_scale(const unsigned* s) {
    return fmaxf(__uint_as_float(*s) * (1.0f / 127.0f), 1e-8f);
}

__global__ void k_init_slots(unsigned* slots) {
    if (threadIdx.x < 16) slots[threadIdx.x] = 0u;
}

__global__ __launch_bounds__(256) void k_absmax(const float* __restrict__ p, size_t n, unsigned* slot) {
    float m = 0.0f;
    for (size_t i = (size_t)blockIdx.x * blockDim.x + threadIdx.x; i < n;
         i += (size_t)gridDim.x * blockDim.x)
        m = fmaxf(m, fabsf(p[i]));
    for (int off = 16; off > 0; off >>= 1) m = fmaxf(m, __shfl_xor(m, off, 32));
    __shared__ float sm[8];
    int lane = threadIdx.x & 31, w = threadIdx.x >> 5;
    if (lane == 0) sm[w] = m;
    __syncthreads();
    if (threadIdx.x == 0) {
        float mm = sm[0];
        for (int i = 1; i < 8; ++i) mm = fmaxf(mm, sm[i]);
        atomicMax(slot, __float_as_uint(mm));
    }
}

__device__ __forceinline__ int8_t q8(float v, float inv_s) {
    float q = rintf(v * inv_s);
    q = fminf(fmaxf(q, -128.0f), 127.0f);
    return (int8_t)(int)q;
}

__global__ __launch_bounds__(256) void k_quant_x_nhwc(const float* __restrict__ x,
                                                      int8_t* __restrict__ qx,
                                                      const unsigned* slot) {
    const float inv = 1.0f / slot_scale(slot);
    const size_t n = (size_t)64 * 28 * 28 * 512;
    for (size_t i = (size_t)blockIdx.x * blockDim.x + threadIdx.x; i < n;
         i += (size_t)gridDim.x * blockDim.x) {
        int c = (int)(i & 511);
        size_t t = i >> 9;
        int w = (int)(t % 28); t /= 28;
        int h = (int)(t % 28);
        int b = (int)(t / 28);
        float v = x[(((size_t)b * 512 + c) * 28 + h) * 28 + w];
        qx[i] = q8(v, inv);
    }
}

__global__ __launch_bounds__(256) void k_quant_w_1x1(const float* __restrict__ w,
                                                     int8_t* __restrict__ qw,
                                                     size_t n, const unsigned* slot) {
    const float inv = 1.0f / slot_scale(slot);
    for (size_t i = (size_t)blockIdx.x * blockDim.x + threadIdx.x; i < n;
         i += (size_t)gridDim.x * blockDim.x)
        qw[i] = q8(w[i], inv);
}

__global__ __launch_bounds__(256) void k_quant_w_3x3(const float* __restrict__ w,
                                                     int8_t* __restrict__ qw,
                                                     const unsigned* slot) {
    const float inv = 1.0f / slot_scale(slot);
    const size_t n = (size_t)256 * 2304;
    for (size_t i = (size_t)blockIdx.x * blockDim.x + threadIdx.x; i < n;
         i += (size_t)gridDim.x * blockDim.x) {
        int o = (int)(i / 2304);
        int r = (int)(i % 2304);
        int c = r / 9;
        int tap = r % 9;
        qw[(size_t)o * 2304 + tap * 256 + c] = q8(w[i], inv);
    }
}

__global__ __launch_bounds__(256) void k_quant_bias(const float* __restrict__ b,
                                                    float* __restrict__ qb,
                                                    int n, const unsigned* slot) {
    const float s = slot_scale(slot);
    const float inv = 1.0f / s;
    for (int i = blockIdx.x * blockDim.x + threadIdx.x; i < n; i += gridDim.x * blockDim.x) {
        float q = fminf(fmaxf(rintf(b[i] * inv), -128.0f), 127.0f);
        qb[i] = q * s;
    }
}

__global__ __launch_bounds__(256) void k_quant_act(const float* __restrict__ a,
                                                   int8_t* __restrict__ q,
                                                   size_t n, const unsigned* slot) {
    const float inv = 1.0f / slot_scale(slot);
    for (size_t i = (size_t)blockIdx.x * blockDim.x + threadIdx.x; i < n;
         i += (size_t)gridDim.x * blockDim.x)
        q[i] = q8(a[i], inv);
}

template <int MODE>
__global__ __launch_bounds__(256) void k_conv_gemm_i8(
    const int8_t* __restrict__ A, const int8_t* __restrict__ W,
    const float* __restrict__ qb,
    const unsigned* __restrict__ sa_slot, const unsigned* __restrict__ sw_slot,
    float* __restrict__ out, unsigned* absmax_slot,
    int M, int N, int K, int do_relu6)
{
    __shared__ __align__(16) int8_t sB[2][64 * 64];

    const int tid = threadIdx.x;
    const int lane = tid & 31;
    const int wv = tid >> 5;
    const int ngrps = N >> 6;
    const int mblk = blockIdx.x / ngrps;
    const int ng = blockIdx.x % ngrps;

    const int mt = mblk * 8 + wv;
    const int mrow = mt * 16 + (lane & 15);
    const int hi = lane >> 4;

    const int8_t* arow = nullptr;
    int ih0 = 0, iw0 = 0;
    size_t pixbase = 0;
    if (MODE == 0) {
        arow = A + (size_t)mrow * K;
    } else if (MODE == 1) {
        int b = mrow / 196, r = mrow % 196;
        int oh = r / 14, ow = r % 14;
        arow = A + ((size_t)(b * 28 + 2 * oh) * 28 + 2 * ow) * (size_t)K;
    } else {
        int b = mrow / 196, r = mrow % 196;
        int oh = r / 14, ow = r % 14;
        ih0 = 2 * oh - 1; iw0 = 2 * ow - 1;
        pixbase = (size_t)b * 784;
    }

    const int sn = tid >> 2;
    const int sp = (tid & 3) * 16;
    const int8_t* wsrc = W + (size_t)(ng * 64 + sn) * K + sp;
    const int nk = K >> 6;

    auto stage = [&](int i) {
        const int8_t* src = wsrc + (size_t)i * 64;
        int8_t* dst = &sB[i & 1][sn * 64 + sp];
#if USE_ASYNC_LDS
        __builtin_amdgcn_global_load_async_to_lds_b128(
            (g_v4i*)(g_i8*)(int8_t*)src, (l_v4i*)(l_i8*)dst, 0, 0);
#else
        *(int4*)dst = *(const int4*)src;
#endif
    };

    v8i acc[4];
#pragma unroll
    for (int t = 0; t < 4; ++t)
#pragma unroll
        for (int r = 0; r < 8; ++r) acc[t][r] = 0;

    const int nl = lane & 15;
    const int kb = hi * 16;

    stage(0);
    for (int i = 0; i < nk; ++i) {
        if (i + 1 < nk) {
            stage(i + 1);
#if USE_ASYNC_LDS
            WAIT_ASYNC(1);
#endif
        } else {
#if USE_ASYNC_LDS
            WAIT_ASYNC(0);
#endif
        }
        __syncthreads();

        const int k0 = i * 64;
        const int8_t* abase;
        bool avalid = true;
        if (MODE == 2) {
            int tap = k0 >> 8;
            int kh = tap / 3, kw = tap - kh * 3;
            int ih = ih0 + kh, iw = iw0 + kw;
            avalid = ((unsigned)ih < 28u) && ((unsigned)iw < 28u);
            abase = A + (pixbase + (size_t)ih * 28 + iw) * 256 + (k0 & 255);
        } else {
            abase = arow + k0;
        }
        v8i a;
#pragma unroll
        for (int v = 0; v < 8; ++v) {
            int koff = 32 * (v >> 2) + 4 * (v & 3) + 16 * hi;
            a[v] = avalid ? *(const int*)(abase + koff) : 0;
        }
#pragma unroll
        for (int t = 0; t < 4; ++t) {
            const int8_t* wb = &sB[i & 1][(t * 16 + nl) * 64 + kb];
            const int4 blo = *(const int4*)(wb);
            const int4 bhi = *(const int4*)(wb + 32);
            v8i b;
            b[0] = blo.x; b[1] = blo.y; b[2] = blo.z; b[3] = blo.w;
            b[4] = bhi.x; b[5] = bhi.y; b[6] = bhi.z; b[7] = bhi.w;
            acc[t] = __builtin_amdgcn_wmma_i32_16x16x64_iu8(
                 true, a,  true, b, acc[t],
                 false,  false);
        }
        __syncthreads();
    }

    const float scale = slot_scale(sa_slot) * slot_scale(sw_slot);
    float amax = 0.0f;
#pragma unroll
    for (int t = 0; t < 4; ++t) {
        const int n = ng * 64 + t * 16 + nl;
        const float bias = qb[n];
#pragma unroll
        for (int r = 0; r < 8; ++r) {
            const int m = mt * 16 + r + 8 * hi;
            float f = (float)acc[t][r] * scale + bias;
            if (do_relu6) f = fminf(fmaxf(f, 0.0f), 6.0f);
            amax = fmaxf(amax, fabsf(f));
            out[(size_t)m * N + n] = f;
        }
    }
    if (absmax_slot) {
        for (int off = 16; off > 0; off >>= 1) amax = fmaxf(amax, __shfl_xor(amax, off, 32));
        if (lane == 0) atomicMax(absmax_slot, __float_as_uint(amax));
    }
}

__global__ __launch_bounds__(256) void k_final(const float* __restrict__ act3,
                                               const float* __restrict__ resid,
                                               const unsigned* slot,
                                               float* __restrict__ out) {
    const float s = slot_scale(slot);
    const float inv = 1.0f / s;
    const size_t n = (size_t)64 * 1024 * 196;
    for (size_t i = (size_t)blockIdx.x * blockDim.x + threadIdx.x; i < n;
         i += (size_t)gridDim.x * blockDim.x) {
        int ow = (int)(i % 14); size_t t = i / 14;
        int oh = (int)(t % 14); t /= 14;
        int c = (int)(t % 1024);
        int b = (int)(t / 1024);
        size_t p = ((size_t)b * 196 + oh * 14 + ow) * 1024 + c;
        float v = act3[p];
        float q = fminf(fmaxf(rintf(v * inv), -128.0f), 127.0f) * s;
        float o = q + resid[p];
        out[i] = fminf(fmaxf(o, 0.0f), 6.0f);
    }
}

extern "C" void kernel_launch(void* const* d_in, const int* in_sizes, int n_in,
                              void* d_out, int out_size, void* d_ws, size_t ws_size,
                              hipStream_t stream) {
    (void)in_sizes; (void)n_in; (void)out_size; (void)ws_size;
    const float* x  = (const float*)d_in[0];
    const float* w1 = (const float*)d_in[1];
    const float* b1 = (const float*)d_in[2];
    const float* w2 = (const float*)d_in[3];
    const float* b2 = (const float*)d_in[4];
    const float* w3 = (const float*)d_in[5];
    const float* b3 = (const float*)d_in[6];
    const float* wsw = (const float*)d_in[7];
    const float* bs  = (const float*)d_in[8];

    uint8_t* base = (uint8_t*)d_ws;
    size_t off = 0;
    auto alloc = [&](size_t bytes) -> void* {
        off = (off + 255) & ~(size_t)255;
        void* p = base + off;
        off += bytes;
        return p;
    };
    unsigned* slots = (unsigned*)alloc(64 * 4);
    float* qb1 = (float*)alloc(256 * 4);
    float* qb2 = (float*)alloc(256 * 4);
    float* qb3 = (float*)alloc(1024 * 4);
    float* qbs = (float*)alloc(1024 * 4);
    int8_t* qw1 = (int8_t*)alloc((size_t)256 * 512);
    int8_t* qw2 = (int8_t*)alloc((size_t)256 * 2304);
    int8_t* qw3 = (int8_t*)alloc((size_t)1024 * 256);
    int8_t* qws = (int8_t*)alloc((size_t)1024 * 512);
    int8_t* qx  = (int8_t*)alloc((size_t)64 * 784 * 512);
    int8_t* qa1 = (int8_t*)alloc((size_t)50176 * 256);
    int8_t* qa2 = (int8_t*)alloc((size_t)12544 * 256);
    float* fa = (float*)alloc((size_t)12544 * 1024 * 4);
    float* fb = (float*)alloc((size_t)12544 * 256 * 4);
    float* fc = (float*)alloc((size_t)12544 * 1024 * 4);

    k_init_slots<<<1, 32, 0, stream>>>(slots);

    k_absmax<<<2048, 256, 0, stream>>>(x,  (size_t)64 * 512 * 784, slots + SL_X);
    k_absmax<<<256, 256, 0, stream>>>(w1, (size_t)131072, slots + SL_W1);
    k_absmax<<<1,   256, 0, stream>>>(b1, (size_t)256,    slots + SL_B1);
    k_absmax<<<512, 256, 0, stream>>>(w2, (size_t)589824, slots + SL_W2);
    k_absmax<<<1,   256, 0, stream>>>(b2, (size_t)256,    slots + SL_B2);
    k_absmax<<<256, 256, 0, stream>>>(w3, (size_t)262144, slots + SL_W3);
    k_absmax<<<4,   256, 0, stream>>>(b3, (size_t)1024,   slots + SL_B3);
    k_absmax<<<512, 256, 0, stream>>>(wsw,(size_t)524288, slots + SL_WS);
    k_absmax<<<4,   256, 0, stream>>>(bs, (size_t)1024,   slots + SL_BS);

    k_quant_x_nhwc<<<8192, 256, 0, stream>>>(x, qx, slots + SL_X);
    k_quant_w_1x1<<<512, 256, 0, stream>>>(w1, qw1, (size_t)131072, slots + SL_W1);
    k_quant_w_3x3<<<1024, 256, 0, stream>>>(w2, qw2, slots + SL_W2);
    k_quant_w_1x1<<<512, 256, 0, stream>>>(w3, qw3, (size_t)262144, slots + SL_W3);
    k_quant_w_1x1<<<1024, 256, 0, stream>>>(wsw, qws, (size_t)524288, slots + SL_WS);
    k_quant_bias<<<1, 256, 0, stream>>>(b1, qb1, 256, slots + SL_B1);
    k_quant_bias<<<1, 256, 0, stream>>>(b2, qb2, 256, slots + SL_B2);
    k_quant_bias<<<4, 256, 0, stream>>>(b3, qb3, 1024, slots + SL_B3);
    k_quant_bias<<<4, 256, 0, stream>>>(bs, qbs, 1024, slots + SL_BS);

    k_conv_gemm_i8<0><<<(50176 / 128) * (256 / 64), 256, 0, stream>>>(
        qx, qw1, qb1, slots + SL_X, slots + SL_W1, fa, slots + SL_A1, 50176, 256, 512, 1);
    k_quant_act<<<8192, 256, 0, stream>>>(fa, qa1, (size_t)50176 * 256, slots + SL_A1);

    k_conv_gemm_i8<2><<<(12544 / 128) * (256 / 64), 256, 0, stream>>>(
        qa1, qw2, qb2, slots + SL_A1, slots + SL_W2, fb, slots + SL_A2, 12544, 256, 2304, 1);
    k_quant_act<<<2048, 256, 0, stream>>>(fb, qa2, (size_t)12544 * 256, slots + SL_A2);

    k_conv_gemm_i8<0><<<(12544 / 128) * (1024 / 64), 256, 0, stream>>>(
        qa2, qw3, qb3, slots + SL_A2, slots + SL_W3, fa, slots + SL_A3, 12544, 1024, 256, 0);

    k_conv_gemm_i8<1><<<(12544 / 128) * (1024 / 64), 256, 0, stream>>>(
        qx, qws, qbs, slots + SL_X, slots + SL_WS, fc, nullptr, 12544, 1024, 512, 0);

    k_final<<<8192, 256, 0, stream>>>(fa, fc, slots + SL_A3, (float*)d_out);
}
